// EdgeGatedConv_86715389706243
// MI455X (gfx1250) — matrix operand order repaired and checked by reading, not compiled, not run on hardware
//
#include <hip/hip_runtime.h>
#include <math.h>

#define NDIM 128

typedef __attribute__((ext_vector_type(16))) __bf16 v16bf;
typedef __attribute__((ext_vector_type(8)))  float  v8f;

__device__ __forceinline__ float sigmoidf_(float x) { return 1.0f / (1.0f + __expf(-x)); }

__device__ __forceinline__ void atomAddF(float* p, float v) {
  __hip_atomic_fetch_add(p, v, __ATOMIC_RELAXED, __HIP_MEMORY_SCOPE_AGENT);
}

// Pack float4 -> 4 bf16 and store as one 8-byte LDS write (ds_store_b64).
__device__ __forceinline__ void store4bf(__bf16* dst, float4 v) {
  union { __bf16 h[4]; uint2 u; } pk;
  pk.h[0] = (__bf16)v.x; pk.h[1] = (__bf16)v.y;
  pk.h[2] = (__bf16)v.z; pk.h[3] = (__bf16)v.w;
  *(uint2*)dst = pk.u;
}

// A fragment (16x32 bf16, wave32) from LDS row-major tile.
// Lanes 0-15: M=lane, elems 0..7 = K kb..kb+7, elems 8..15 = K kb+16..kb+23
// Lanes 16-31: M=lane-16, K offset +8 (per ISA 16-bit A layout).
__device__ __forceinline__ v16bf a_frag_lds(const __bf16* tile, int stride, int kbase, int lane) {
  const __bf16* row = tile + (lane & 15) * stride + kbase + ((lane >> 4) << 3);
  v16bf a;
#pragma unroll
  for (int e = 0; e < 8; ++e) { a[e] = row[e]; a[e + 8] = row[e + 16]; }
  return a;
}

// B fragment from pre-swizzled packed weights: each (kt, ntile, lane) owns 16
// contiguous bf16 (32 bytes, 32B-aligned) -> two global_load_b128 per fragment.
__device__ __forceinline__ v16bf b_frag_pk(const __bf16* __restrict__ P,
                                           int kt, int nt, int lane) {
  return *(const v16bf*)(P + ((((kt << 3) + nt) << 5) + lane) * 16);
}

// ---------------------------------------------------------------------------
// Weight pre-pack: f32 [K x 128] row-major -> bf16 B-fragment layout
// P[((kt*8 + nt)*32 + lane)*16 + e] = W[(kt*32 + (lane>>4)*8 + (e&7) + (e>>3)*16)*128 + nt*16 + (lane&15)]
// ---------------------------------------------------------------------------
__global__ void k_pack(const float* __restrict__ W, __bf16* __restrict__ P, int KT) {
  const int total = KT * 4096;
  for (int i = blockIdx.x * blockDim.x + threadIdx.x; i < total;
       i += gridDim.x * blockDim.x) {
    int e    = i & 15;
    int lane = (i >> 4) & 31;
    int nt   = (i >> 9) & 7;
    int kt   = i >> 12;
    int k = kt * 32 + ((lane >> 4) << 3) + (e & 7) + ((e >> 3) << 4);
    int n = nt * 16 + (lane & 15);
    P[i] = (__bf16)W[k * NDIM + n];
  }
}

__device__ __forceinline__ void prefetch_weights(const __bf16* P, int bytes, int tid) {
  for (int i = tid * 128; i < bytes; i += 256 * 128)
    __builtin_prefetch((const char*)P + i, 0, 3);
}

// ---------------------------------------------------------------------------
// Kernel 1: LineGraphConv message GEMM + scatter-add (sum & count).
// 16 line-edges per block, 256 threads = 8 waves; wave w owns output cols
// [16w,16w+16). K = 257 = 8 x 32 (WMMA) + rank-1 scalar column.
// ---------------------------------------------------------------------------
__global__ void __launch_bounds__(256)
k_line(const float* __restrict__ lg_x, const float* __restrict__ lg_ea,
       const float* __restrict__ Wp, const float* __restrict__ W_line,
       const float* __restrict__ b_line,
       const int* __restrict__ lg_src, const int* __restrict__ lg_dst,
       float* __restrict__ lgsum, float* __restrict__ cnt) {
  __shared__ __bf16 sA[16 * 264];
  __shared__ int sDst[16];
  const int tid = threadIdx.x;
  const long base = (long)blockIdx.x * 16;
  const __bf16* P = (const __bf16*)Wp;

  prefetch_weights(P, 8 * 4096 * 2, tid);

  // Gather: thread (r = tid>>4) loads 8 src + 8 dst floats as float4s.
  {
    const int r = tid >> 4;
    const long le = base + r;
    const int s = lg_src[le];
    const int d = lg_dst[le];
    const float4* ps = (const float4*)(lg_x + (long)s * NDIM);
    const float4* pd = (const float4*)(lg_x + (long)d * NDIM);
    const int q = (tid & 15) * 2;           // float4 index within row
    __bf16* rowA = sA + r * 264;
    store4bf(rowA + q * 4,           ps[q]);
    store4bf(rowA + q * 4 + 4,       ps[q + 1]);
    store4bf(rowA + 128 + q * 4,     pd[q]);
    store4bf(rowA + 128 + q * 4 + 4, pd[q + 1]);
    if ((tid & 15) == 0) rowA[256] = (__bf16)lg_ea[le];
  }
  if (tid < 16) {
    int d = lg_dst[base + tid];
    sDst[tid] = d;
    atomAddF(&cnt[d], 1.0f);
  }
  __syncthreads();

  const int lane = tid & 31;
  const int nt = tid >> 5;
  const int n = (nt << 4) + (lane & 15);
  v8f acc = {};
#pragma unroll
  for (int kt = 0; kt < 8; ++kt) {
    v16bf a = a_frag_lds(sA, 264, kt * 32, lane);
    v16bf b = b_frag_pk(P, kt, nt, lane);
    acc = __builtin_amdgcn_wmma_f32_16x16x32_bf16(false, a, false, b,
                                                  (short)0, acc, false, false);
  }
  const float w256 = W_line[256 * NDIM + n];  // rank-1 tail for K index 256
  const float bn = b_line[n];
  const int mh = (lane >> 4) << 3;
#pragma unroll
  for (int r = 0; r < 8; ++r) {
    int m = r + mh;
    float v = acc[r] + (float)sA[m * 264 + 256] * w256 + bn;
    v = v * sigmoidf_(v);  // SiLU
    atomAddF(&lgsum[(long)sDst[m] * NDIM + n], v);
  }
}

// ---------------------------------------------------------------------------
// Kernel 2: finalize scatter-mean -> lg_x_new; residual + LayerNorm ->
// edge_attr_new; gate GEMM (sigmoid); gated atomic scatter into atom agg.
// ---------------------------------------------------------------------------
__global__ void __launch_bounds__(256)
k_edge(const float* __restrict__ edge_attr, const float* __restrict__ lg_x,
       const float* __restrict__ Wp, const float* __restrict__ b_gate,
       const float* __restrict__ g_edge, const float* __restrict__ beta_edge,
       const int* __restrict__ col_idx, const float* __restrict__ cnt,
       float* __restrict__ lg_io,     // in: segment sum, out: lg_x_new
       float* __restrict__ edge_out,  // edge_attr_new
       float* __restrict__ agg_atom)  // x-region accumulator (zeroed)
{
  __shared__ __bf16 sA[16 * 136];
  __shared__ float  sE[16 * NDIM];
  __shared__ int    sCol[16];
  const int tid = threadIdx.x;
  const long base = (long)blockIdx.x * 16;
  const __bf16* P = (const __bf16*)Wp;

  prefetch_weights(P, 4 * 4096 * 2, tid);

  // Row-parallel LayerNorm: row = tid>>4, 8 contiguous cols per thread;
  // the 16 threads of a row are a 16-lane shuffle group inside the wave.
  const int row = tid >> 4;
  const int c0 = (tid & 15) * 8;
  const long e = base + row;
  const float inv = 1.0f / fmaxf(cnt[e], 1.0f);
  float v[8];
  float s1 = 0.0f, s2 = 0.0f;
#pragma unroll
  for (int j = 0; j < 8; ++j) {
    long off = e * NDIM + c0 + j;
    float lgn = lg_x[off] + lg_io[off] * inv;   // scatter-mean + residual
    lg_io[off] = lgn;                           // lg_x_new output
    float t = edge_attr[off] + lgn;
    v[j] = t; s1 += t; s2 += t * t;
  }
#pragma unroll
  for (int m = 1; m < 16; m <<= 1) {
    s1 += __shfl_xor(s1, m, 16);
    s2 += __shfl_xor(s2, m, 16);
  }
  const float mean = s1 * (1.0f / NDIM);
  const float var = s2 * (1.0f / NDIM) - mean * mean;
  const float rstd = rsqrtf(var + 1e-5f);
  float y[8];
#pragma unroll
  for (int j = 0; j < 8; ++j) {
    int c = c0 + j;
    y[j] = (v[j] - mean) * rstd * g_edge[c] + beta_edge[c];
    edge_out[e * NDIM + c] = y[j];
    sE[row * NDIM + c] = y[j];
  }
  store4bf(sA + row * 136 + c0,     make_float4(y[0], y[1], y[2], y[3]));
  store4bf(sA + row * 136 + c0 + 4, make_float4(y[4], y[5], y[6], y[7]));
  if (tid < 16) sCol[tid] = col_idx[base + tid];
  __syncthreads();

  const int lane = tid & 31;
  const int nt = tid >> 5;
  const int n = (nt << 4) + (lane & 15);
  v8f acc = {};
#pragma unroll
  for (int kt = 0; kt < 4; ++kt) {
    v16bf a = a_frag_lds(sA, 136, kt * 32, lane);
    v16bf b = b_frag_pk(P, kt, nt, lane);
    acc = __builtin_amdgcn_wmma_f32_16x16x32_bf16(false, a, false, b,
                                                  (short)0, acc, false, false);
  }
  const float bn = b_gate[n];
  const int mh = (lane >> 4) << 3;
#pragma unroll
  for (int r = 0; r < 8; ++r) {
    int m = r + mh;
    float gate = sigmoidf_(acc[r] + bn);
    atomAddF(&agg_atom[(long)sCol[m] * NDIM + n], gate * sE[m * NDIM + n]);
  }
}

// ---------------------------------------------------------------------------
// Kernel 3: AtomGraphConv: concat(x, agg) [16x256] GEMM + SiLU + residual +
// LayerNorm -> x_new (written in place over the agg accumulator region).
// ---------------------------------------------------------------------------
__global__ void __launch_bounds__(256)
k_atom(const float* __restrict__ x, const float* __restrict__ Wp,
       const float* __restrict__ b_atom, const float* __restrict__ g_node,
       const float* __restrict__ beta_node,
       float* __restrict__ x_io)  // in: agg (accumulated), out: x_new
{
  __shared__ __bf16 sA[16 * 264];
  __shared__ float  sH[16 * NDIM];
  const int tid = threadIdx.x;
  const long base = (long)blockIdx.x * 16;
  const __bf16* P = (const __bf16*)Wp;

  prefetch_weights(P, 8 * 4096 * 2, tid);

  // Gather: thread (r = tid>>4) loads 8 x-floats + 8 agg-floats as float4s.
  {
    const int r = tid >> 4;
    const long node = base + r;
    const float4* px = (const float4*)(x + node * NDIM);
    const float4* pa = (const float4*)(x_io + node * NDIM);
    const int q = (tid & 15) * 2;
    __bf16* rowA = sA + r * 264;
    store4bf(rowA + q * 4,           px[q]);
    store4bf(rowA + q * 4 + 4,       px[q + 1]);
    store4bf(rowA + 128 + q * 4,     pa[q]);
    store4bf(rowA + 128 + q * 4 + 4, pa[q + 1]);
  }
  __syncthreads();

  const int lane = tid & 31;
  const int nt = tid >> 5;
  const int n = (nt << 4) + (lane & 15);
  v8f acc = {};
#pragma unroll
  for (int kt = 0; kt < 8; ++kt) {
    v16bf a = a_frag_lds(sA, 264, kt * 32, lane);
    v16bf b = b_frag_pk(P, kt, nt, lane);
    acc = __builtin_amdgcn_wmma_f32_16x16x32_bf16(false, a, false, b,
                                                  (short)0, acc, false, false);
  }
  const float bn = b_atom[n];
  const int mh = (lane >> 4) << 3;
#pragma unroll
  for (int r = 0; r < 8; ++r) {
    int m = r + mh;
    float h = acc[r] + bn;
    h = h * sigmoidf_(h);                            // SiLU
    sH[m * NDIM + n] = x[(base + m) * NDIM + n] + h; // residual pre-norm
  }
  __syncthreads();

  const int row = tid >> 4;
  const int c0 = (tid & 15) * 8;
  float v[8];
  float s1 = 0.0f, s2 = 0.0f;
#pragma unroll
  for (int j = 0; j < 8; ++j) {
    float t = sH[row * NDIM + c0 + j];
    v[j] = t; s1 += t; s2 += t * t;
  }
#pragma unroll
  for (int m = 1; m < 16; m <<= 1) {
    s1 += __shfl_xor(s1, m, 16);
    s2 += __shfl_xor(s2, m, 16);
  }
  const float mean = s1 * (1.0f / NDIM);
  const float var = s2 * (1.0f / NDIM) - mean * mean;
  const float rstd = rsqrtf(var + 1e-5f);
#pragma unroll
  for (int j = 0; j < 8; ++j) {
    int c = c0 + j;
    x_io[(base + row) * NDIM + c] = (v[j] - mean) * rstd * g_node[c] + beta_node[c];
  }
}

extern "C" void kernel_launch(void* const* d_in, const int* in_sizes, int n_in,
                              void* d_out, int out_size, void* d_ws, size_t ws_size,
                              hipStream_t stream) {
  const float* x         = (const float*)d_in[0];
  const float* edge_attr = (const float*)d_in[1];
  const float* lg_x      = (const float*)d_in[2];
  const float* lg_ea     = (const float*)d_in[3];
  const float* W_line    = (const float*)d_in[4];
  const float* b_line    = (const float*)d_in[5];
  const float* W_gate    = (const float*)d_in[6];
  const float* b_gate    = (const float*)d_in[7];
  const float* W_atom    = (const float*)d_in[8];
  const float* b_atom    = (const float*)d_in[9];
  const float* g_node    = (const float*)d_in[10];
  const float* beta_node = (const float*)d_in[11];
  const float* g_edge    = (const float*)d_in[12];
  const float* beta_edge = (const float*)d_in[13];
  const int*   eidx      = (const int*)d_in[14];   // [2, E]
  const int*   lgidx     = (const int*)d_in[15];   // [2, LE]

  const long N  = in_sizes[0] / NDIM;   // 200000
  const long E  = in_sizes[1] / NDIM;   // 1000000
  const long LE = in_sizes[3];          // 1000000

  float* x_out    = (float*)d_out;          // x_new; doubles as atom agg accumulator
  float* edge_out = x_out + N * NDIM;       // edge_attr_new
  float* lg_out   = edge_out + E * NDIM;    // lg_x_new; doubles as segment-sum accumulator

  float*  cnt    = (float*)d_ws;                          // E floats
  __bf16* p_line = (__bf16*)((char*)d_ws + (size_t)E * 4); // 8*4096 bf16
  __bf16* p_gate = p_line + 8 * 4096;                      // 4*4096 bf16
  __bf16* p_atom = p_gate + 4 * 4096;                      // 8*4096 bf16

  hipMemsetAsync(x_out, 0, (size_t)N * NDIM * sizeof(float), stream);
  hipMemsetAsync(lg_out, 0, (size_t)E * NDIM * sizeof(float), stream);
  hipMemsetAsync(cnt, 0, (size_t)E * sizeof(float), stream);

  dim3 blk(256);
  k_pack<<<dim3(32), blk, 0, stream>>>(W_line, p_line, 8);
  k_pack<<<dim3(32), blk, 0, stream>>>(W_gate, p_gate, 4);
  k_pack<<<dim3(32), blk, 0, stream>>>(W_atom, p_atom, 8);

  k_line<<<dim3((unsigned)(LE / 16)), blk, 0, stream>>>(
      lg_x, lg_ea, (const float*)p_line, W_line, b_line,
      lgidx, lgidx + LE, lg_out, cnt);
  k_edge<<<dim3((unsigned)(E / 16)), blk, 0, stream>>>(
      edge_attr, lg_x, (const float*)p_gate, b_gate, g_edge, beta_edge,
      eidx + E, cnt, lg_out, edge_out, x_out);
  k_atom<<<dim3((unsigned)(N / 16)), blk, 0, stream>>>(
      x, (const float*)p_atom, b_atom, g_node, beta_node, x_out);
}
